// StockLSTM_2448131359199
// MI455X (gfx1250) — compile-verified, matrix operand order checked by reading, not run on hardware
//
#include <hip/hip_runtime.h>
#include <hip/hip_bf16.h>

typedef __attribute__((ext_vector_type(16))) __bf16 v16bf;
typedef __attribute__((ext_vector_type(8)))  float  v8f;
typedef __attribute__((ext_vector_type(2)))  float  v2f;

#define T_LEN 512
#define IN_D  8
#define HID   64
#define NGATE 256   // 4*HID
#define ROWS  16    // batch rows per block
#define HOFF  (ROWS * HID)

__device__ __forceinline__ v8f wmma_bf16(v16bf a, v16bf b, v8f c) {
    return __builtin_amdgcn_wmma_f32_16x16x32_bf16(
        false, a, false, b, (short)0, c, false, false);
}
__device__ __forceinline__ v8f wmma_f32k4(v2f a, v2f b, v8f c) {
    return __builtin_amdgcn_wmma_f32_16x16x4_f32(
        false, a, false, b, (short)0, c, false, false);
}

// bf16 16x32 (A) / 32x16 (B) fragment from a 64-element bf16 row:
// halves 0-7 = row[32*ks + kb .. +7], halves 8-15 = row[32*ks + kb+16 .. +23]
__device__ __forceinline__ v16bf load_frag(const __bf16* row, int ks, int kb) {
    union { uint4 u[2]; v16bf v; } f;
    const uint4* p = reinterpret_cast<const uint4*>(row + 32 * ks + kb);
    f.u[0] = p[0];
    f.u[1] = p[2];   // +16 bf16 elements = +32 bytes
    return f.v;
}

// f32 16x4 (A) / 4x16 (B) fragment from an 8-element f32 row (K=8 = 2 ksteps)
__device__ __forceinline__ v2f load_frag_f32(const float* row, int ks, int koff) {
    union { double d; v2f v; } f;
    f.d = *reinterpret_cast<const double*>(row + 4 * ks + koff);
    return f.v;
}

// 8 hardware tanh (TRANS32) in one block: each op is the required independent
// instruction for the previous one's result hazard; v_nop covers the last.
__device__ __forceinline__ void tanh8_hw(float* t) {
    asm volatile(
        "v_tanh_f32 %0, %0\n\t"
        "v_tanh_f32 %1, %1\n\t"
        "v_tanh_f32 %2, %2\n\t"
        "v_tanh_f32 %3, %3\n\t"
        "v_tanh_f32 %4, %4\n\t"
        "v_tanh_f32 %5, %5\n\t"
        "v_tanh_f32 %6, %6\n\t"
        "v_tanh_f32 %7, %7\n\t"
        "v_nop"
        : "+v"(t[0]), "+v"(t[1]), "+v"(t[2]), "+v"(t[3]),
          "+v"(t[4]), "+v"(t[5]), "+v"(t[6]), "+v"(t[7]));
}

// Gate nonlinearities + cell update for one layer (all tiles share C layout).
// pb[g] = 0.5*bias for sigmoid gates (i,f,o), full bias for the tanh gate (g).
__device__ __forceinline__ void cell_update(v8f* acc, const float* pb, float* c,
                                            __bf16* hw_base, int sbase) {
    float ti[8], tf[8], tg[8], to[8], tc[8];
    #pragma unroll
    for (int v = 0; v < 8; ++v) {
        ti[v] = fmaf(acc[0][v], 0.5f, pb[0]);   // 0.5*(x+b) for sigmoid-as-tanh
        tf[v] = fmaf(acc[1][v], 0.5f, pb[1]);
        tg[v] = acc[2][v] + pb[2];
        to[v] = fmaf(acc[3][v], 0.5f, pb[3]);
    }
    tanh8_hw(ti); tanh8_hw(tf); tanh8_hw(tg); tanh8_hw(to);
    #pragma unroll
    for (int v = 0; v < 8; ++v) {
        const float iv = fmaf(ti[v], 0.5f, 0.5f);
        const float fv = fmaf(tf[v], 0.5f, 0.5f);
        c[v] = fmaf(fv, c[v], iv * tg[v]);
        tc[v] = c[v];
    }
    tanh8_hw(tc);
    #pragma unroll
    for (int v = 0; v < 8; ++v) {
        const float ov = fmaf(to[v], 0.5f, 0.5f);
        hw_base[sbase + v * HID] = (__bf16)(ov * tc[v]);
    }
}

__global__ __launch_bounds__(128)
void lstm2_fused_kernel(const float* __restrict__ x,
                        const float* __restrict__ Wih0, const float* __restrict__ Whh0,
                        const float* __restrict__ b0,
                        const float* __restrict__ Wih1, const float* __restrict__ Whh1,
                        const float* __restrict__ b1,
                        const float* __restrict__ fcw,  const float* __restrict__ fcb,
                        float* __restrict__ out)
{
    extern __shared__ char smem[];
    __bf16* whh0 = (__bf16*)smem;                 // [256][64] bf16 (read once into regs)
    __bf16* wih1 = whh0 + NGATE * HID;            // [256][64] bf16
    __bf16* whh1 = wih1 + NGATE * HID;            // [256][64] bf16
    float*  xl   = (float*)(whh1 + NGATE * HID);  // [16][8] f32
    __bf16* hA   = (__bf16*)(xl + ROWS * IN_D);   // [2][16][64] bf16 (double buffer)
    __bf16* hB   = hA + 2 * HOFF;                 // [2][16][64] bf16

    const int tid  = threadIdx.x;
    const int lane = tid & 31;
    const int wv   = tid >> 5;                    // wave id 0..3
    const int brow = blockIdx.x * ROWS;

    // ---- one-time: stage bf16 weights into LDS, zero h buffers, preload x0 ----
    for (int i = tid; i < NGATE * HID; i += 128) {
        whh0[i] = (__bf16)Whh0[i];
        wih1[i] = (__bf16)Wih1[i];
        whh1[i] = (__bf16)Whh1[i];
    }
    for (int i = tid; i < 2 * HOFF; i += 128) {
        hA[i] = (__bf16)0.0f; hB[i] = (__bf16)0.0f;
    }
    const float* gx = x + (long)(brow + (tid >> 3)) * (T_LEN * IN_D) + (tid & 7);
    xl[tid] = gx[0];
    __syncthreads();

    const int N    = lane & 15;          // C/D layout: N = lane%16
    const int Mb   = (lane >> 4) * 8;    // C/D layout: M = Mb + v
    const int kb   = Mb;                 // bf16 frag K-base for this lane half
    const int koff = (lane >> 4) * 2;    // f32  frag K-base for this lane half

    // ---- loop-invariant register-resident operands ----
    v2f   wI0[4][2];                     // Wih0^T f32 fragments (from global)
    v16bf wB0[4][2];                     // Whh0^T bf16 fragments (from LDS)
    const __bf16* w1row[4];              // layer-1 B-frag row pointers (LDS)
    const __bf16* w2row[4];
    float pb0[4], pb1[4];                // folded biases (0.5*b for i,f,o; b for g)
    #pragma unroll
    for (int g = 0; g < 4; ++g) {
        const int col = 16 * (4 * g + wv) + N;       // wave-local gate interleave
        const int r   = (16 * (4 * g + wv) + N) * HID;
        wI0[g][0] = load_frag_f32(Wih0 + col * IN_D, 0, koff);
        wI0[g][1] = load_frag_f32(Wih0 + col * IN_D, 1, koff);
        wB0[g][0] = load_frag(whh0 + r, 0, kb);
        wB0[g][1] = load_frag(whh0 + r, 1, kb);
        w1row[g]  = wih1 + r;
        w2row[g]  = whh1 + r;
        const float bz0 = b0[col], bz1 = b1[col];
        pb0[g] = (g == 2) ? bz0 : 0.5f * bz0;
        pb1[g] = (g == 2) ? bz1 : 0.5f * bz1;
    }
    const float* xrow = xl + N * IN_D;
    const int arow  = N * HID;                       // A-frag row offset within buffer
    const int sbase = Mb * HID + 16 * wv + N;        // h store base offset

    float cA[8], cB[8];
    #pragma unroll
    for (int v = 0; v < 8; ++v) { cA[v] = 0.0f; cB[v] = 0.0f; }

    for (int t = 0; t < T_LEN; ++t) {
        const int rb = t & 1, wb = rb ^ 1;

        // ================= layer 0 =================
        v8f acc[4] = {};                 // first WMMA uses inline-0 C operand
        { // x_t @ Wih0^T via f32 WMMA (exact), weights in registers
            const v2f xa0 = load_frag_f32(xrow, 0, koff);
            const v2f xa1 = load_frag_f32(xrow, 1, koff);
            #pragma unroll
            for (int g = 0; g < 4; ++g) {
                acc[g] = wmma_f32k4(xa0, wI0[g][0], acc[g]);
                acc[g] = wmma_f32k4(xa1, wI0[g][1], acc[g]);
            }
        }
        { // + hA(t-1) @ Whh0^T via bf16 WMMA, weights in registers
            const v16bf a0 = load_frag(hA + rb * HOFF + arow, 0, kb);
            const v16bf a1 = load_frag(hA + rb * HOFF + arow, 1, kb);
            #pragma unroll
            for (int g = 0; g < 4; ++g) {
                acc[g] = wmma_bf16(a0, wB0[g][0], acc[g]);
                acc[g] = wmma_bf16(a1, wB0[g][1], acc[g]);
            }
        }
        cell_update(acc, pb0, cA, hA + wb * HOFF, sbase);
        __syncthreads();

        // ================= layer 1 =================
        v8f acd[4] = {};
        {
            const v16bf aA0 = load_frag(hA + wb * HOFF + arow, 0, kb);
            const v16bf aA1 = load_frag(hA + wb * HOFF + arow, 1, kb);
            const v16bf aB0 = load_frag(hB + rb * HOFF + arow, 0, kb);
            const v16bf aB1 = load_frag(hB + rb * HOFF + arow, 1, kb);
            #pragma unroll
            for (int g = 0; g < 4; ++g) {
                acd[g] = wmma_bf16(aA0, load_frag(w1row[g], 0, kb), acd[g]);
                acd[g] = wmma_bf16(aA1, load_frag(w1row[g], 1, kb), acd[g]);
                acd[g] = wmma_bf16(aB0, load_frag(w2row[g], 0, kb), acd[g]);
                acd[g] = wmma_bf16(aB1, load_frag(w2row[g], 1, kb), acd[g]);
            }
        }
        cell_update(acd, pb1, cB, hB + wb * HOFF, sbase);

        // stage next x_t into LDS + prefetch t+2 (old xl readers done pre-mid-barrier)
        if (t + 1 < T_LEN) xl[tid] = gx[(t + 1) * IN_D];
        if (t + 2 < T_LEN) __builtin_prefetch(&gx[(t + 2) * IN_D], 0, 1);
        __syncthreads();
    }

    // ---- final FC: out[b] = hB_last[b,:] . fcw + fcb (T even -> buffer 0) ----
    if (tid < ROWS) {
        const __bf16* hf = hB;
        float s = fcb[0];
        #pragma unroll 8
        for (int j = 0; j < HID; ++j) s += (float)hf[tid * HID + j] * fcw[j];
        out[brow + tid] = s;
    }
}

extern "C" void kernel_launch(void* const* d_in, const int* in_sizes, int n_in,
                              void* d_out, int out_size, void* d_ws, size_t ws_size,
                              hipStream_t stream) {
    const float* x    = (const float*)d_in[0];
    const float* Wih0 = (const float*)d_in[1];
    const float* Whh0 = (const float*)d_in[2];
    const float* b0   = (const float*)d_in[3];
    const float* Wih1 = (const float*)d_in[4];
    const float* Whh1 = (const float*)d_in[5];
    const float* b1   = (const float*)d_in[6];
    const float* fcw  = (const float*)d_in[7];
    const float* fcb  = (const float*)d_in[8];
    float* out = (float*)d_out;

    const size_t lds =
        3 * NGATE * HID * sizeof(__bf16)   // bf16 weights (96 KB)
      + ROWS * IN_D * sizeof(float)        // x tile
      + 4 * HOFF * sizeof(__bf16);         // hA/hB double buffers

    const int nblocks = 1024 / ROWS;       // 64 blocks, 16 batch rows each
    lstm2_fused_kernel<<<nblocks, 128, lds, stream>>>(
        x, Wih0, Whh0, b0, Wih1, Whh1, b1, fcw, fcb, out);
}
